// GPTBigCodeBlock_58050777973167
// MI455X (gfx1250) — matrix operand order repaired and checked by reading, not compiled, not run on hardware
//
#include <hip/hip_runtime.h>
#include <math.h>

// Problem constants (match reference)
#define BB   64
#define KVN  4096
#define DD   2048
#define HH   16
#define HDD  128
#define PASTW (2*HDD)            // 256 floats per (b,kv) row
#define QKVN (DD + 2*HDD)        // 2304
#define FFN  (4*DD)              // 8192
#define MASK_VALUE (-3.4028234663852886e38f)
#define PS 0.08838834764831845f  // 1/sqrt(128) == ds*unscale

typedef __attribute__((ext_vector_type(16))) __bf16 v16bf;
typedef __attribute__((ext_vector_type(8)))  float  v8f;
typedef __attribute__((ext_vector_type(4)))  unsigned int u32x4;
typedef __attribute__((ext_vector_type(8)))  int i32x8;
typedef __attribute__((ext_vector_type(4)))  int i32x4;

union Frag {
  v16bf bf;
  u32x4 q[2];
  unsigned int u[8];
};

// ---------------------------------------------------------------------------
// bf16 conversion helpers.
// Hot path: round-half-up + one v_perm_b32 splicing the two high halves.
// ---------------------------------------------------------------------------
__device__ __forceinline__ unsigned int pk2bf(float x, float y) {
  unsigned int ux = __float_as_uint(x) + 0x8000u;
  unsigned int uy = __float_as_uint(y) + 0x8000u;
  // out bytes {3,2}=uy{3,2}, {1,0}=ux{3,2}
  return __builtin_amdgcn_perm(uy, ux, 0x07060302u);
}
// RNE scalar (epilogues / one-shot weight pack)
__device__ __forceinline__ unsigned short f2bf(float x) {
  unsigned int u = __float_as_uint(x);
  u = (u + 0x7FFFu + ((u >> 16) & 1u)) >> 16;
  return (unsigned short)u;
}
__device__ __forceinline__ unsigned int pk2bf_rne(float x, float y) {
  return (unsigned int)f2bf(x) | ((unsigned int)f2bf(y) << 16);
}

__device__ __forceinline__ v8f wmma_bf16(const Frag& a, const Frag& b, v8f c) {
  // D = A(16x32 bf16) * B(32x16 bf16) + C(16x16 f32)
  return __builtin_amdgcn_wmma_f32_16x16x32_bf16(
      false, a.bf, false, b.bf, (short)0, c, false, false);
}

// ---------------------------------------------------------------------------
// Tensor Data Mover: 1-D contiguous tile (ndw dwords) global -> LDS.
// D# built per CDNA5 ISA:
//   g0: count=1 | lds_addr | global_addr[56:0] | type=2
//   g1: data_size=4B; tensor_dim0=ndw; tensor_dim1=1; tile_dim0=ndw
// ---------------------------------------------------------------------------
#if defined(__gfx1250__) && __has_builtin(__builtin_amdgcn_tensor_load_to_lds) && \
    __has_builtin(__builtin_amdgcn_s_wait_tensorcnt)
#define USE_TDM 1
#if __has_include(<hip/amd_detail/amd_gfx1250_TDM.h>)
#define TDM_ARITY6 1
#else
#define TDM_ARITY6 0
#endif
#else
#define USE_TDM 0
#endif

#if USE_TDM
__device__ __forceinline__ void tdm_load_1d(unsigned int lds_byte_addr,
                                            const void* gptr, unsigned int ndw) {
  unsigned long long ga = (unsigned long long)(uintptr_t)gptr;
  u32x4 g0 = {0u, 0u, 0u, 0u};
  g0[0] = 1u;                                         // count = 1 valid descriptor
  g0[1] = lds_byte_addr;                              // lds_addr
  g0[2] = (unsigned int)ga;                           // global_addr[31:0]
  g0[3] = (unsigned int)((ga >> 32) & 0x01FFFFFFu) | 0x80000000u;  // [56:32] | type=2
  i32x8 g1 = {0, 0, 0, 0, 0, 0, 0, 0};
  g1[0] = (int)(2u << 16);                            // data_size = 4 bytes
  g1[1] = (int)((ndw & 0xFFFFu) << 16);               // tensor_dim0[15:0]
  g1[2] = (int)((ndw >> 16) | (1u << 16));            // tensor_dim0[31:16] | tensor_dim1=1
  g1[3] = (int)((ndw & 0xFFFFu) << 16);               // tile_dim0 = ndw (1-D tile)
  i32x4 z4 = {0, 0, 0, 0};
#if TDM_ARITY6
  i32x8 z8 = {0, 0, 0, 0, 0, 0, 0, 0};
  __builtin_amdgcn_tensor_load_to_lds(g0, g1, z4, z4, z8, 0);
#else
  __builtin_amdgcn_tensor_load_to_lds(g0, g1, z4, z4, 0);
#endif
}
#endif

// ---------------------------------------------------------------------------
// Weight pack: fp32 row-major W[K][N] -> bf16 in WMMA B-fragment tile order.
// Tile (nt,kt): 256 dwords, lane-major (lane*8 + v). Per ISA B layout:
//   VGPR v, lanes 0-15 : N = lane,    K = 2v, 2v+1 (v<4); 2v+8, 2v+9 (v>=4)
//   VGPR v, lanes 16-31: N = lane-16, K = 8+2v, 9+2v (v<4); 16+2v, 17+2v (v>=4)
// ---------------------------------------------------------------------------
__global__ void pack_w(const float* __restrict__ W, unsigned int* __restrict__ Wp,
                       int K, int N) {
  size_t idx = (size_t)blockIdx.x * 256 + threadIdx.x;
  size_t total = ((size_t)K * N) >> 1;
  if (idx >= total) return;
  int ktiles = K >> 5;
  size_t tile = idx >> 8;
  int r = (int)(idx & 255);
  int lane = r >> 3, v = r & 7;
  int nt = (int)(tile / ktiles), kt = (int)(tile % ktiles);
  int n = (nt << 4) + (lane & 15);
  int k = (kt << 5) + (v < 4 ? 2 * v : 16 + 2 * (v - 4)) + ((lane >= 16) ? 8 : 0);
  Wp[idx] = pk2bf_rne(W[(size_t)k * N + n], W[(size_t)(k + 1) * N + n]);
}

// ---------------------------------------------------------------------------
// Fused residual-add + LayerNorm -> bf16 activations (row-major).
// ---------------------------------------------------------------------------
__global__ __launch_bounds__(256) void ln_fuse(
    const float* __restrict__ x, const float* __restrict__ y,
    const float* __restrict__ w, const float* __restrict__ b,
    float* __restrict__ resid_out, unsigned short* __restrict__ out_bf) {
  __shared__ float s1[256], s2[256];
  const int row = blockIdx.x, t = threadIdx.x;
  const float* xr = x + (size_t)row * DD;
  const float* yr = y ? (y + (size_t)row * DD) : nullptr;
  float vals[8], sum = 0.f, sq = 0.f;
#pragma unroll
  for (int i = 0; i < 8; ++i) {
    int j = t + (i << 8);
    float v = xr[j] + (yr ? yr[j] : 0.f);
    vals[i] = v; sum += v; sq += v * v;
  }
  s1[t] = sum; s2[t] = sq; __syncthreads();
  for (int o = 128; o > 0; o >>= 1) {
    if (t < o) { s1[t] += s1[t + o]; s2[t] += s2[t + o]; }
    __syncthreads();
  }
  const float mean = s1[0] * (1.f / DD);
  const float var  = s2[0] * (1.f / DD) - mean * mean;
  const float rinv = rsqrtf(var + 1e-5f);
#pragma unroll
  for (int i = 0; i < 8; ++i) {
    int j = t + (i << 8);
    if (resid_out) resid_out[(size_t)row * DD + j] = vals[i];
    out_bf[(size_t)row * DD + j] = f2bf((vals[i] - mean) * rinv * w[j] + b[j]);
  }
}

// ---------------------------------------------------------------------------
// Generic bf16 WMMA GEMM: C(64 x N) = A(64 x K, bf16 row-major) * Wp + bias.
// grid = (N/16, 4), block = 32 (one wave = one 16x16 output tile).
// Weight tiles stream through LDS via the Tensor Data Mover, double-buffered
// in 4 KB chunks (4 k-steps); A-fragments come straight from VMEM.
// MODE: 0 store f32; 1 +resid, store f32 (in place ok); 2 gelu->bf16; 3 f32.
// ---------------------------------------------------------------------------
#define CHUNK_KT 4               // k-steps per TDM chunk (4*256 dwords = 4 KB)

template <int MODE>
__global__ __launch_bounds__(32) void gemm16(
    const unsigned int* __restrict__ A, const unsigned int* __restrict__ Wp,
    const float* __restrict__ bias, int K, int N,
    float* __restrict__ outF, unsigned short* __restrict__ outB,
    const float* __restrict__ resid) {
  const int lane = threadIdx.x & 31;
  const int mi = lane & 15;
  const bool hi = lane >= 16;
  const int nt = blockIdx.x;
  const int m0 = blockIdx.y << 4;
  const int ksteps = K >> 5;
  const unsigned int* arow = A + (size_t)(m0 + mi) * (K >> 1);
  const unsigned int* wbase = Wp + (size_t)nt * ksteps * 256;
  v8f c = {};

#if USE_TDM
  __shared__ unsigned int wtile[2][CHUNK_KT * 256];
  const unsigned int lds0 = (unsigned int)(uintptr_t)&wtile[0][0];
  const unsigned int lds1 = (unsigned int)(uintptr_t)&wtile[1][0];
  const int nch = ksteps / CHUNK_KT;

  auto consume = [&](int cc) {
    const unsigned int* lw = &wtile[cc & 1][0] + lane * 8;
#pragma unroll
    for (int s = 0; s < CHUNK_KT; ++s) {
      Frag a, b;
      const int kt = cc * CHUNK_KT + s;
      const int o = (kt << 4) + (hi ? 4 : 0);
      a.q[0] = *(const u32x4*)(arow + o);
      a.q[1] = *(const u32x4*)(arow + o + 8);
      b.q[0] = *(const u32x4*)(lw + s * 256);          // ds_load_b128
      b.q[1] = *(const u32x4*)(lw + s * 256 + 4);
      c = wmma_bf16(a, b, c);
    }
  };

  tdm_load_1d(lds0, wbase, CHUNK_KT * 256);
  for (int cc = 0; cc < nch - 1; ++cc) {
    tdm_load_1d(((cc + 1) & 1) ? lds1 : lds0,
                wbase + (size_t)(cc + 1) * CHUNK_KT * 256, CHUNK_KT * 256);
    __builtin_amdgcn_s_wait_tensorcnt(1);              // chunk cc resident
    consume(cc);
  }
  __builtin_amdgcn_s_wait_tensorcnt(0);                // last chunk resident
  consume(nch - 1);
#else
  const unsigned int* wptr = wbase + lane * 8;
  for (int kt = 0; kt < ksteps; ++kt) {
    Frag a, b;
    const int o = (kt << 4) + (hi ? 4 : 0);
    a.q[0] = *(const u32x4*)(arow + o);
    a.q[1] = *(const u32x4*)(arow + o + 8);
    b.q[0] = *(const u32x4*)(wptr);
    b.q[1] = *(const u32x4*)(wptr + 4);
    __builtin_prefetch(wptr + 1024, 0, 1);
    wptr += 256;
    c = wmma_bf16(a, b, c);
  }
#endif

  const int n = (nt << 4) + mi;
#pragma unroll
  for (int i = 0; i < 8; ++i) {
    const int m = m0 + i + (hi ? 8 : 0);
    float v = c[i] + bias[n];
    if (MODE == 0) {
      outF[(size_t)m * N + n] = v;
    } else if (MODE == 1) {
      v += resid[(size_t)m * N + n];
      outF[(size_t)m * N + n] = v;
    } else if (MODE == 2) {
      float g = 0.5f * v * (1.f + tanhf(0.7978845608028654f * (v + 0.044715f * v * v * v)));
      outB[(size_t)m * N + n] = f2bf(g);
    } else {
      outF[(size_t)m * N + n] = v;
    }
  }
}

// ---------------------------------------------------------------------------
// layer_past copy (float4 wide) + single-row KV fixup.
// ---------------------------------------------------------------------------
__global__ void copy4(const float4* __restrict__ s, float4* __restrict__ d, size_t n) {
  size_t i = (size_t)blockIdx.x * blockDim.x + threadIdx.x;
  if (i < n) d[i] = s[i];
}
__global__ void kv_fixup(const float* __restrict__ qkv, float* __restrict__ past,
                         const int* __restrict__ keylen) {
  int b = blockIdx.x, j = threadIdx.x;                   // j in [0,256)
  int row = keylen[0] - 1;
  past[((size_t)b * KVN + row) * PASTW + j] = qkv[(size_t)b * QKVN + DD + j];
}

// ---------------------------------------------------------------------------
// Scores: per batch b, S(16 heads x 4096 kv) = Q_b(16x128) @ K_b^T * 1/sqrt(HD),
// masked. One block per batch, 8 waves stride the 256 kv-tiles. K=128 -> 4
// WMMA k-steps; Q fragment built once (fp32 -> bf16 in-register).
// ---------------------------------------------------------------------------
__global__ __launch_bounds__(256) void attn_scores(
    const float* __restrict__ qkv, const float* __restrict__ past,
    const unsigned char* __restrict__ mask, float* __restrict__ scores) {
  const int b = blockIdx.x;
  const int wave = threadIdx.x >> 5;
  const int lane = threadIdx.x & 31;
  const int head = lane & 15;
  const bool hi = lane >= 16;
  const float* qb = qkv + (size_t)b * QKVN;              // first DD = queries
  Frag afr[4];
#pragma unroll
  for (int kt = 0; kt < 4; ++kt)
#pragma unroll
    for (int v = 0; v < 8; ++v) {
      int d = (kt << 5) + (v < 4 ? 2 * v : 16 + 2 * (v - 4)) + (hi ? 8 : 0);
      afr[kt].u[v] = pk2bf(qb[head * HDD + d], qb[head * HDD + d + 1]);
    }
  const float* kc = past + (size_t)b * KVN * PASTW;      // K at [kv*256 + d]
  const unsigned char* mrow = mask + (size_t)b * KVN;
  for (int nt = wave; nt < (KVN >> 4); nt += 8) {
    const int kv = (nt << 4) + (lane & 15);
    const float* kr = kc + (size_t)kv * PASTW;
    v8f c = {};
#pragma unroll
    for (int kt = 0; kt < 4; ++kt) {
      Frag bfr;
#pragma unroll
      for (int v = 0; v < 8; ++v) {
        int d = (kt << 5) + (v < 4 ? 2 * v : 16 + 2 * (v - 4)) + (hi ? 8 : 0);
        float2 kk = *(const float2*)(kr + d);            // one b64 load per pair
        bfr.u[v] = pk2bf(kk.x, kk.y);
      }
      c = wmma_bf16(afr[kt], bfr, c);
    }
#pragma unroll
    for (int i = 0; i < 8; ++i) {
      int h = i + (hi ? 8 : 0);
      float s = c[i] * PS;
      if (!mrow[kv]) s = MASK_VALUE;
      scores[((size_t)b * HH + h) * KVN + kv] = s;
    }
  }
}

// Row softmax over 4096, output bf16 probs (row-major per (b,h)).
__global__ __launch_bounds__(256) void softmax_row(
    const float* __restrict__ scores, unsigned short* __restrict__ probs) {
  __shared__ float red[256];
  const int row = blockIdx.x, t = threadIdx.x;
  const float* sr = scores + (size_t)row * KVN;
  float vals[16], mx = -INFINITY;
#pragma unroll
  for (int i = 0; i < 16; ++i) { vals[i] = sr[t + (i << 8)]; mx = fmaxf(mx, vals[i]); }
  red[t] = mx; __syncthreads();
  for (int o = 128; o > 0; o >>= 1) { if (t < o) red[t] = fmaxf(red[t], red[t + o]); __syncthreads(); }
  mx = red[0]; __syncthreads();
  float sum = 0.f;
#pragma unroll
  for (int i = 0; i < 16; ++i) { vals[i] = __expf(vals[i] - mx); sum += vals[i]; }
  red[t] = sum; __syncthreads();
  for (int o = 128; o > 0; o >>= 1) { if (t < o) red[t] += red[t + o]; __syncthreads(); }
  const float rs = 1.f / red[0];
#pragma unroll
  for (int i = 0; i < 16; ++i) probs[(size_t)row * KVN + t + (i << 8)] = f2bf(vals[i] * rs);
}

// attn_b(16x128) = P_b(16x4096, bf16) @ V_b(4096x128, fp32->bf16 in-register).
// One block per batch; 8 waves = 8 d-tiles of 16. 128 WMMA k-steps.
__global__ __launch_bounds__(256) void attn_v(
    const unsigned int* __restrict__ probs_dw, const float* __restrict__ past,
    unsigned short* __restrict__ attn_bf) {
  const int b = blockIdx.x;
  const int wv = threadIdx.x >> 5;
  const int lane = threadIdx.x & 31;
  const bool hi = lane >= 16;
  const unsigned int* prow = probs_dw + ((size_t)b * HH + (lane & 15)) * (KVN >> 1);
  const float* vc = past + (size_t)b * KVN * PASTW + HDD;  // V at [kv*256 + 128 + d]
  const int d = (wv << 4) + (lane & 15);
  v8f c = {};
  for (int kt = 0; kt < (KVN >> 5); ++kt) {
    Frag a, bb;
    const int o = (kt << 4) + (hi ? 4 : 0);
    a.q[0] = *(const u32x4*)(prow + o);
    a.q[1] = *(const u32x4*)(prow + o + 8);
#pragma unroll
    for (int v = 0; v < 8; ++v) {
      int kv = (kt << 5) + (v < 4 ? 2 * v : 16 + 2 * (v - 4)) + (hi ? 8 : 0);
      bb.u[v] = pk2bf(vc[(size_t)kv * PASTW + d], vc[(size_t)(kv + 1) * PASTW + d]);
    }
    c = wmma_bf16(a, bb, c);
  }
#pragma unroll
  for (int i = 0; i < 8; ++i) {
    int h = i + (hi ? 8 : 0);
    attn_bf[(size_t)b * DD + h * HDD + d] = f2bf(c[i]);
  }
}

// ---------------------------------------------------------------------------
extern "C" void kernel_launch(void* const* d_in, const int* in_sizes, int n_in,
                              void* d_out, int out_size, void* d_ws, size_t ws_size,
                              hipStream_t stream) {
  const float* hs_in = (const float*)d_in[0];
  const float* res_in = (const float*)d_in[1];
  const float* past_in = (const float*)d_in[2];
  const unsigned char* mask = (const unsigned char*)d_in[3];
  const int* keylen = (const int*)d_in[4];
  const float* aaw = (const float*)d_in[5];
  const float* aab = (const float*)d_in[6];
  const float* apw = (const float*)d_in[7];
  const float* apb = (const float*)d_in[8];
  const float* l1w = (const float*)d_in[9];
  const float* l1b = (const float*)d_in[10];
  const float* l2w = (const float*)d_in[11];
  const float* l2b = (const float*)d_in[12];
  const float* mfw = (const float*)d_in[13];
  const float* mfb = (const float*)d_in[14];
  const float* mpw = (const float*)d_in[15];
  const float* mpb = (const float*)d_in[16];

  float* out_hs = (float*)d_out;                    // (64,2048)
  float* out_res = out_hs + (size_t)BB * DD;        // (64,2048)
  float* out_past = out_res + (size_t)BB * DD;      // (64,4096,256)

  char* w = (char*)d_ws;
  auto alloc = [&](size_t bytes) -> void* {
    void* p = (void*)w;
    w += (bytes + 255) & ~(size_t)255;
    return p;
  };
  unsigned int* aaw_p = (unsigned int*)alloc((size_t)DD * QKVN * 2);
  unsigned int* apw_p = (unsigned int*)alloc((size_t)DD * DD * 2);
  unsigned int* mfw_p = (unsigned int*)alloc((size_t)DD * FFN * 2);
  unsigned int* mpw_p = (unsigned int*)alloc((size_t)FFN * DD * 2);
  unsigned short* hs1_bf = (unsigned short*)alloc((size_t)BB * DD * 2);
  unsigned short* hs2_bf = (unsigned short*)alloc((size_t)BB * DD * 2);
  unsigned short* attn_bf = (unsigned short*)alloc((size_t)BB * DD * 2);
  unsigned short* act_bf = (unsigned short*)alloc((size_t)BB * FFN * 2);
  float* qkv = (float*)alloc((size_t)BB * QKVN * 4);
  float* scores = (float*)alloc((size_t)BB * HH * KVN * 4);
  unsigned short* probs = (unsigned short*)alloc((size_t)BB * HH * KVN * 2);

  // 1) pack weights fp32 -> bf16 WMMA tile layout
  pack_w<<<(unsigned)(((size_t)DD * QKVN / 2 + 255) / 256), 256, 0, stream>>>(aaw, aaw_p, DD, QKVN);
  pack_w<<<(unsigned)(((size_t)DD * DD / 2 + 255) / 256), 256, 0, stream>>>(apw, apw_p, DD, DD);
  pack_w<<<(unsigned)(((size_t)DD * FFN / 2 + 255) / 256), 256, 0, stream>>>(mfw, mfw_p, DD, FFN);
  pack_w<<<(unsigned)(((size_t)FFN * DD / 2 + 255) / 256), 256, 0, stream>>>(mpw, mpw_p, FFN, DD);

  // 2) residual add + LN1 (resid -> out_res, normalized bf16 -> hs1_bf)
  ln_fuse<<<BB, 256, 0, stream>>>(hs_in, res_in, l1w, l1b, out_res, hs1_bf);

  // 3) QKV GEMM (f32 out)
  gemm16<0><<<dim3(QKVN / 16, 4), 32, 0, stream>>>(
      (const unsigned int*)hs1_bf, aaw_p, aab, DD, QKVN, qkv, nullptr, nullptr);

  // 4) layer_past copy + single-row fixup
  copy4<<<(unsigned)(((size_t)BB * KVN * PASTW / 4 + 255) / 256), 256, 0, stream>>>(
      (const float4*)past_in, (float4*)out_past, (size_t)BB * KVN * PASTW / 4);
  kv_fixup<<<BB, PASTW, 0, stream>>>(qkv, out_past, keylen);

  // 5) attention
  attn_scores<<<BB, 256, 0, stream>>>(qkv, out_past, mask, scores);
  softmax_row<<<BB * HH, 256, 0, stream>>>(scores, probs);
  attn_v<<<BB, 256, 0, stream>>>((const unsigned int*)probs, out_past, attn_bf);

  // 6) attn projection + residual (in place on out_res)
  gemm16<1><<<dim3(DD / 16, 4), 32, 0, stream>>>(
      (const unsigned int*)attn_bf, apw_p, apb, DD, DD, out_res, nullptr, out_res);

  // 7) LN2 -> hs2_bf
  ln_fuse<<<BB, 256, 0, stream>>>(out_res, nullptr, l2w, l2b, nullptr, hs2_bf);

  // 8) MLP fc + gelu -> bf16
  gemm16<2><<<dim3(FFN / 16, 4), 32, 0, stream>>>(
      (const unsigned int*)hs2_bf, mfw_p, mfb, DD, FFN, nullptr, act_bf, nullptr);

  // 9) MLP proj -> out_hs
  gemm16<3><<<dim3(DD / 16, 4), 32, 0, stream>>>(
      (const unsigned int*)act_bf, mpw_p, mpb, FFN, DD, out_hs, nullptr, nullptr);
}
